// NearestNeighborTokenizer_39659728011469
// MI455X (gfx1250) — matrix operand order repaired and checked by reading, not compiled, not run on hardware
//
#include <hip/hip_runtime.h>
#include <stdint.h>

// ---------------------------------------------------------------------------
// NearestNeighborTokenizer on MI455X (gfx1250, wave32, WMMA)
//
// d2[t][n] = ||x_t||^2 + ||c_n||^2 - 2 * <x_t, c_n>, clamp >= 0
// out[t]   = (min_n d2 <= 100) ? argmin_n d2 : -1
//
// Cross term via V_WMMA_F32_16X16X32_BF16 with a 3-product hi/lo bf16 split
// (xh*ch + xh*cl + xl*ch) for ~fp32 accuracy at matrix-core rates.
// Codes' hi/lo bf16 split is precomputed once (pass 1) so the hot loop is
// pure fragment loads + WMMA, no per-tile conversion VALU.
// ---------------------------------------------------------------------------

typedef __attribute__((ext_vector_type(16))) __bf16 v16bf;
typedef __attribute__((ext_vector_type(8)))  float  v8f;

#define NTOK        2048      // 2 * 1024 tokens
#define DIM         64
#define NCODES      50000
#define NTILE_CODES 3125      // 50000 / 16
#define CSPLIT      8         // code-stripe blocks (blockIdx.y)
#define WPB         8         // waves per block (256 threads, wave32)
#define THRESH      100.0f

// workspace layout (byte offsets)
#define OFF_BEST    0u                      // u64[2048]   = 16384 B
#define OFF_X2      16384u                  // f32[2048]   =  8192 B
#define OFF_C2      24576u                  // f32[50000]  = 200000 B
#define OFF_CHI     224576u                 // bf16[50000*64] = 6400000 B
#define OFF_CLO     6624576u                // bf16[50000*64] = 6400000 B
#define WS_FAST     13024576u
#define WS_MIN      224576u

// ---------------- pass 1: row norms, init, optional code hi/lo split --------
__global__ void nn_prep_kernel(const float* __restrict__ x,
                               const float* __restrict__ codes,
                               unsigned long long* __restrict__ best,
                               float* __restrict__ x2,
                               float* __restrict__ c2,
                               __bf16* __restrict__ c_hi,
                               __bf16* __restrict__ c_lo,
                               int write_hilo) {
    int t = blockIdx.x * blockDim.x + threadIdx.x;
    if (t < NTOK) {
        const float4* p = (const float4*)(x + (size_t)t * DIM);
        float s = 0.0f;
        #pragma unroll
        for (int i = 0; i < DIM / 4; ++i) {
            float4 v = p[i];
            s += v.x * v.x + v.y * v.y + v.z * v.z + v.w * v.w;
        }
        x2[t] = s;
        best[t] = ~0ull;
    }
    if (t < NCODES) {
        const float4* p = (const float4*)(codes + (size_t)t * DIM);
        float s = 0.0f;
        #pragma unroll
        for (int i = 0; i < DIM / 4; ++i) {
            float4 v = p[i];
            s += v.x * v.x + v.y * v.y + v.z * v.z + v.w * v.w;
            if (write_hilo) {
                float fe[4] = {v.x, v.y, v.z, v.w};
                #pragma unroll
                for (int j = 0; j < 4; ++j) {
                    __bf16 h = (__bf16)fe[j];
                    c_hi[(size_t)t * DIM + 4 * i + j] = h;
                    c_lo[(size_t)t * DIM + 4 * i + j] = (__bf16)(fe[j] - (float)h);
                }
            }
        }
        c2[t] = s;
    }
}

// Build per-wave A fragments (hi + residual) for one 16-token tile.
// Lane layout (16-bit A 16x32, contiguous K to match B): lanes 0-15 hold M=lane,
// K = {0..15}; lanes 16-31 hold M=lane-16, K = {16..31}.
__device__ __forceinline__ void load_a_frags(const float* __restrict__ x,
                                             int tokBase, int half, int l16,
                                             v16bf ah[2], v16bf al[2]) {
    const float* r = x + (size_t)(tokBase + l16) * DIM;
    #pragma unroll
    for (int c = 0; c < 2; ++c) {
        const int kA0 = c * 32 + half * 16;       // elements e0..e7
        const int kA1 = c * 32 + half * 16 + 8;   // elements e8..e15
        v16bf h, l;
        #pragma unroll
        for (int e = 0; e < 8; ++e) {
            float f0 = r[kA0 + e];
            float f1 = r[kA1 + e];
            __bf16 h0 = (__bf16)f0;
            __bf16 h1 = (__bf16)f1;
            h[e]     = h0;
            h[e + 8] = h1;
            l[e]     = (__bf16)(f0 - (float)h0);
            l[e + 8] = (__bf16)(f1 - (float)h1);
        }
        ah[c] = h;
        al[c] = l;
    }
}

// Shared epilogue: pack (d2,id) keys, reduce across each 16-lane half, atomic.
__device__ __forceinline__ void reduce_and_commit(float bestd[8], int bestid[8],
                                                  int tokBase, int half, int l16,
                                                  unsigned long long* __restrict__ best) {
    unsigned long long bst[8];
    #pragma unroll
    for (int v = 0; v < 8; ++v) {
        // d2 >= 0 -> float bits order-preserve as u32; id in low bits gives
        // lowest-index tie-break like jnp.argmin.
        bst[v] = ((unsigned long long)__float_as_uint(bestd[v]) << 32) |
                 (unsigned)bestid[v];
    }
    #pragma unroll
    for (int m = 1; m <= 8; m <<= 1) {
        #pragma unroll
        for (int v = 0; v < 8; ++v) {
            unsigned long long o = __shfl_xor(bst[v], m, 32);
            bst[v] = o < bst[v] ? o : bst[v];
        }
    }
    if (l16 == 0) {
        #pragma unroll
        for (int v = 0; v < 8; ++v)
            atomicMin(&best[tokBase + half * 8 + v], bst[v]);
    }
}

// ---------------- pass 2 (fast): precomputed bf16 code fragments ------------
__launch_bounds__(256)
__global__ void nn_main_fast(const float* __restrict__ x,
                             const __bf16* __restrict__ c_hi,
                             const __bf16* __restrict__ c_lo,
                             const float* __restrict__ x2,
                             const float* __restrict__ c2,
                             unsigned long long* __restrict__ best) {
    const int lane = threadIdx.x & 31;
    const int wave = threadIdx.x >> 5;
    const int half = lane >> 4;
    const int l16  = lane & 15;
    const int tokBase = blockIdx.x * 16;

    v16bf ah[2], al[2];
    load_a_frags(x, tokBase, half, l16, ah, al);

    float myx2[8];
    #pragma unroll
    for (int v = 0; v < 8; ++v) myx2[v] = x2[tokBase + half * 8 + v];

    float bestd[8];
    int   bestid[8];
    #pragma unroll
    for (int v = 0; v < 8; ++v) { bestd[v] = 3.4e38f; bestid[v] = 0; }

    // stride over 16-code tiles: uniform per wave -> EXEC all-ones at WMMA
    const int stripe = blockIdx.y * WPB + wave;
    for (int tile = stripe; tile < NTILE_CODES; tile += CSPLIT * WPB) {
        const int n = tile * 16 + l16;               // this lane's code column
        const __bf16* crh = c_hi + (size_t)n * DIM;
        const __bf16* crl = c_lo + (size_t)n * DIM;
        const float cc2 = c2[n];

        v8f acc = {};
        #pragma unroll
        for (int c = 0; c < 2; ++c) {
            // B fragment (32x16 bf16): lane holds N=l16, 16 consecutive K
            // (lanes 0-15: K=c*32..+15, lanes 16-31: K=c*32+16..+31) --
            // 32B-aligned contiguous runs, loaded straight from workspace.
            const int kB = c * 32 + half * 16;
            v16bf bh = *(const v16bf*)(crh + kB);
            v16bf bl = *(const v16bf*)(crl + kB);
            acc = __builtin_amdgcn_wmma_f32_16x16x32_bf16(
                false, ah[c], false, bh, (short)0, acc, false, false);
            acc = __builtin_amdgcn_wmma_f32_16x16x32_bf16(
                false, ah[c], false, bl, (short)0, acc, false, false);
            acc = __builtin_amdgcn_wmma_f32_16x16x32_bf16(
                false, al[c], false, bh, (short)0, acc, false, false);
        }

        #pragma unroll
        for (int v = 0; v < 8; ++v) {
            float d2 = fmaxf(myx2[v] + cc2 - 2.0f * acc[v], 0.0f);
            bool lt = d2 < bestd[v];        // strict: keeps lowest n on ties
            bestd[v]  = lt ? d2 : bestd[v];
            bestid[v] = lt ? n  : bestid[v];
        }
    }

    reduce_and_commit(bestd, bestid, tokBase, half, l16, best);
}

// ---------------- pass 2 (fallback): convert codes in-loop ------------------
__launch_bounds__(256)
__global__ void nn_main_conv(const float* __restrict__ x,
                             const float* __restrict__ codes,
                             const float* __restrict__ x2,
                             const float* __restrict__ c2,
                             unsigned long long* __restrict__ best) {
    const int lane = threadIdx.x & 31;
    const int wave = threadIdx.x >> 5;
    const int half = lane >> 4;
    const int l16  = lane & 15;
    const int tokBase = blockIdx.x * 16;

    v16bf ah[2], al[2];
    load_a_frags(x, tokBase, half, l16, ah, al);

    float myx2[8];
    #pragma unroll
    for (int v = 0; v < 8; ++v) myx2[v] = x2[tokBase + half * 8 + v];

    float bestd[8];
    int   bestid[8];
    #pragma unroll
    for (int v = 0; v < 8; ++v) { bestd[v] = 3.4e38f; bestid[v] = 0; }

    const int stripe = blockIdx.y * WPB + wave;
    for (int tile = stripe; tile < NTILE_CODES; tile += CSPLIT * WPB) {
        const int n = tile * 16 + l16;
        const float* cr = codes + (size_t)n * DIM;
        const float cc2 = c2[n];

        v8f acc = {};
        #pragma unroll
        for (int c = 0; c < 2; ++c) {
            const int kB = c * 32 + half * 16;
            const float4* cr4 = (const float4*)(cr + kB);
            v16bf bh, bl;
            #pragma unroll
            for (int q = 0; q < 4; ++q) {
                float4 fv = cr4[q];
                float fe[4] = {fv.x, fv.y, fv.z, fv.w};
                #pragma unroll
                for (int j = 0; j < 4; ++j) {
                    __bf16 h = (__bf16)fe[j];
                    bh[4 * q + j] = h;
                    bl[4 * q + j] = (__bf16)(fe[j] - (float)h);
                }
            }
            acc = __builtin_amdgcn_wmma_f32_16x16x32_bf16(
                false, ah[c], false, bh, (short)0, acc, false, false);
            acc = __builtin_amdgcn_wmma_f32_16x16x32_bf16(
                false, ah[c], false, bl, (short)0, acc, false, false);
            acc = __builtin_amdgcn_wmma_f32_16x16x32_bf16(
                false, al[c], false, bh, (short)0, acc, false, false);
        }

        #pragma unroll
        for (int v = 0; v < 8; ++v) {
            float d2 = fmaxf(myx2[v] + cc2 - 2.0f * acc[v], 0.0f);
            bool lt = d2 < bestd[v];
            bestd[v]  = lt ? d2 : bestd[v];
            bestid[v] = lt ? n  : bestid[v];
        }
    }

    reduce_and_commit(bestd, bestid, tokBase, half, l16, best);
}

// ---------------- pass 3: decode + threshold --------------------------------
__global__ void nn_finalize_kernel(const unsigned long long* __restrict__ best,
                                   int* __restrict__ out) {
    int t = blockIdx.x * blockDim.x + threadIdx.x;
    if (t < NTOK) {
        unsigned long long b = best[t];
        float d2 = __uint_as_float((unsigned)(b >> 32));
        int   id = (int)(unsigned)(b & 0xFFFFFFFFull);
        out[t] = (d2 <= THRESH) ? id : -1;
    }
}

// ---------------------------------------------------------------------------
extern "C" void kernel_launch(void* const* d_in, const int* in_sizes, int n_in,
                              void* d_out, int out_size, void* d_ws, size_t ws_size,
                              hipStream_t stream) {
    const float* x     = (const float*)d_in[0];   // [2,1024,64] f32
    const float* codes = (const float*)d_in[1];   // [50000,64]  f32

    unsigned long long* best = (unsigned long long*)((char*)d_ws + OFF_BEST);
    float*  x2   = (float*) ((char*)d_ws + OFF_X2);
    float*  c2   = (float*) ((char*)d_ws + OFF_C2);
    __bf16* c_hi = (__bf16*)((char*)d_ws + OFF_CHI);
    __bf16* c_lo = (__bf16*)((char*)d_ws + OFF_CLO);

    const int use_fast = (ws_size >= (size_t)WS_FAST) ? 1 : 0;  // ws_size is
    // fixed across calls -> deterministic path selection.

    nn_prep_kernel<<<(NCODES + 255) / 256, 256, 0, stream>>>(
        x, codes, best, x2, c2,
        use_fast ? c_hi : nullptr, use_fast ? c_lo : nullptr, use_fast);

    dim3 grid(NTOK / 16, CSPLIT);   // 128 token tiles x 8 code stripes
    if (use_fast) {
        nn_main_fast<<<grid, 256, 0, stream>>>(x, c_hi, c_lo, x2, c2, best);
    } else {
        nn_main_conv<<<grid, 256, 0, stream>>>(x, codes, x2, c2, best);
    }

    // reference output dtype is int32 (indices / -1 sentinel)
    nn_finalize_kernel<<<(NTOK + 255) / 256, 256, 0, stream>>>(best, (int*)d_out);
}
